// MultiHeadAttention_52106543235430
// MI455X (gfx1250) — matrix operand order repaired and checked by reading, not compiled, not run on hardware
//
#include <hip/hip_runtime.h>
#include <hip/hip_bf16.h>

// ---------------------------------------------------------------------------
// MHA forward for gfx1250 (MI455X): bf16 WMMA pipeline, f32 softmax/accum.
// K-deep GEMMs stage 64x128 B tiles in LDS via the Tensor Data Mover
// (double-buffered, s_wait_tensorcnt), shared by all 4 waves of the block;
// A fragments are issued before the stage barrier so their latency is hidden.
// ---------------------------------------------------------------------------

typedef __bf16 bf16;
typedef __attribute__((ext_vector_type(16))) __bf16 v16bf;
typedef __attribute__((ext_vector_type(8)))  __bf16 v8bf;
typedef __attribute__((ext_vector_type(8)))  float  v8f;
typedef __attribute__((ext_vector_type(4)))  float  v4f;
typedef __attribute__((ext_vector_type(4)))  unsigned int v4u;
typedef __attribute__((ext_vector_type(8)))  int v8i;
typedef __attribute__((ext_vector_type(4)))  int v4i;

#define E_  1024
#define D_  64
#define H_  16
#define B_  4
#define S_  2048
#define HD_ 1024
#define KS_ 128   // K depth of one LDS B-stage

#if defined(__has_builtin)
#  if __has_builtin(__builtin_amdgcn_tensor_load_to_lds)
#    define USE_TDM 1
#  endif
#endif
#ifndef USE_TDM
#  define USE_TDM 0
#endif
#if __has_include(<hip/amd_detail/amd_gfx1250_TDM.h>)
#  define TDM_6ARG 1
#endif

// ---- fragment helpers (CDNA5 wave32 WMMA 16x16x32 bf16 layouts) ------------

static __device__ __forceinline__ v16bf cat16(v8bf lo, v8bf hi) {
  return __builtin_shufflevector(lo, hi, 0,1,2,3,4,5,6,7,8,9,10,11,12,13,14,15);
}
static __device__ __forceinline__ v8bf ld8(const bf16* p) {
  return *(const v8bf*)p;  // 16B load
}

// A fragment (16x32, M x K): lanes 0-15 hold K {kc+0..7, kc+16..23},
// lanes 16-31 hold K {kc+8..15, kc+24..31}; rowPtr points at (row, kc).
static __device__ __forceinline__ v16bf loadA(const bf16* rowPtr, int lane) {
  const bf16* p = rowPtr + ((lane >> 4) << 3);
  return cat16(ld8(p), ld8(p + 16));
}

// B fragment (32x16, K x N): lanes 0-15 hold K {kc+0..7, kc+16..23},
// lanes 16-31 hold K {kc+8..15, kc+24..31};
// colPtr points at (col, kc) of a K-contiguous (transposed) layout.
static __device__ __forceinline__ v16bf loadB(const bf16* colPtr, int lane) {
  const bf16* p = colPtr + ((lane >> 4) << 3);
  return cat16(ld8(p), ld8(p + 16));
}

// A fragment built from f32 data (for softmax weights), converted to bf16.
static __device__ __forceinline__ v16bf loadA_f32(const float* rowPtr, int lane) {
  const float* p = rowPtr + ((lane >> 4) << 3);
  v4f a0 = *(const v4f*)(p);
  v4f a1 = *(const v4f*)(p + 4);
  v4f b0 = *(const v4f*)(p + 16);
  v4f b1 = *(const v4f*)(p + 20);
  v16bf r;
#pragma unroll
  for (int i = 0; i < 4; ++i) {
    r[i]      = (bf16)a0[i];
    r[4 + i]  = (bf16)a1[i];
    r[8 + i]  = (bf16)b0[i];
    r[12 + i] = (bf16)b1[i];
  }
  return r;
}

static __device__ __forceinline__ v8f wmma_bf16(v16bf a, v16bf b, v8f c) {
  // (neg_a, A, neg_b, B, c_mod, C, reuse_a, reuse_b)
  return __builtin_amdgcn_wmma_f32_16x16x32_bf16(false, a, false, b, (short)0, c,
                                                 false, false);
}

// ---- Tensor Data Mover: 2D tile (bf16) global -> LDS -----------------------
// D# packing per CDNA5 ISA ch.8: group0 {count, lds_addr, global_addr, type=2},
// group1 {data_size=2B, tensor_dim0/1, tile_dim0/1, tensor_dim0_stride}.

#if USE_TDM
static __device__ __forceinline__ void tdm_load_tile_2d(
    unsigned lds_off, const void* gptr,
    unsigned tile_d0, unsigned tile_d1,
    unsigned tensor_d0, unsigned tensor_d1, unsigned stride0_elems) {
  unsigned long long ga = (unsigned long long)(__SIZE_TYPE__)gptr;
  v4u g0;
  g0[0] = 1u;                                                // count=1 (valid)
  g0[1] = lds_off;                                           // lds_addr (bytes)
  g0[2] = (unsigned)(ga & 0xFFFFFFFFu);                      // global_addr lo
  g0[3] = (unsigned)((ga >> 32) & 0x01FFFFFFu) | (2u << 30); // addr hi | type=2
  v8i g1;
  g1[0] = (int)(1u << 16);                                   // data_size=2B
  g1[1] = (int)((tensor_d0 & 0xFFFFu) << 16);                // td0[15:0]
  g1[2] = (int)(((tensor_d0 >> 16) & 0xFFFFu) | ((tensor_d1 & 0xFFFFu) << 16));
  g1[3] = (int)(((tensor_d1 >> 16) & 0xFFFFu) | ((tile_d0 & 0xFFFFu) << 16));
  g1[4] = (int)(tile_d1 & 0xFFFFu);                          // tile_dim1, dim2=0
  g1[5] = (int)stride0_elems;                                // td0_stride[31:0]
  g1[6] = 0;                                                 // stride hi, td1s lo
  g1[7] = 0;
  v4i z4 = {};
#if defined(TDM_6ARG)
  v8i z8 = {};
  __builtin_amdgcn_tensor_load_to_lds(g0, g1, z4, z4, z8, 0);
#else
  __builtin_amdgcn_tensor_load_to_lds(g0, g1, z4, z4, 0);
#endif
}
#endif

// Stage one 64 x KS_ bf16 B tile (rows = output cols, K contiguous) into LDS.
static __device__ __forceinline__ void stage_B(bf16* dst, const bf16* src,
                                               unsigned ld, int tidx, int wave) {
#if USE_TDM
  if (wave == 0)
    tdm_load_tile_2d((unsigned)(__SIZE_TYPE__)dst, src,
                     /*tile*/KS_, 64, /*tensor*/ld, 64, /*stride0*/ld);
#else
  (void)wave;
  int n = tidx >> 1, half = (tidx & 1) * 64;
#pragma unroll
  for (int i = 0; i < 8; ++i)
    *(v8bf*)&dst[n * KS_ + half + i * 8] = ld8(src + (size_t)n * ld + half + i * 8);
#endif
}

static __device__ __forceinline__ void stage_wait(int wave, bool more_pending) {
#if USE_TDM
  if (wave == 0) {
    if (more_pending) __builtin_amdgcn_s_wait_tensorcnt(1);
    else              __builtin_amdgcn_s_wait_tensorcnt(0);
  }
#else
  (void)wave; (void)more_pending;
#endif
}

// ---- conversion / transpose kernels ---------------------------------------

__global__ void k_cvt_x(const float* __restrict__ x, bf16* __restrict__ xb, int n) {
  int i = blockIdx.x * 256 + threadIdx.x;
  if (i < n) xb[i] = (bf16)x[i];
}

// Wq/Wk/Wv: [H,E,D] f32 -> [H,D,E] bf16 (K-contiguous for B fragments)
__global__ void k_cvt_w_qkv(const float* __restrict__ W, bf16* __restrict__ Wt) {
  int i = blockIdx.x * 256 + threadIdx.x;      // over H*E*D
  int d = i % D_;
  int e = (i / D_) % E_;
  int h = i / (D_ * E_);
  Wt[((size_t)h * D_ + d) * E_ + e] = (bf16)W[i];
}

// Wo: [HD,E] f32 -> [E,HD] bf16
__global__ void k_cvt_wo(const float* __restrict__ Wo, bf16* __restrict__ Wot) {
  int i = blockIdx.x * 256 + threadIdx.x;      // over HD*E
  int e = i % E_;
  int k = i / E_;
  Wot[(size_t)e * HD_ + k] = (bf16)Wo[i];
}

// ---- QKV projection: per (proj, h, b), GEMM M=S, N=D, K=E ------------------
// grid (S/64, H*B, 3), block 128 (4 waves); wave = 16 rows x 64 cols.
// B (weight) tiles staged in LDS via TDM, double-buffered, shared by 4 waves.

__global__ void k_qkv(const bf16* __restrict__ xb,
                      const bf16* __restrict__ Wqt,
                      const bf16* __restrict__ Wkt,
                      const bf16* __restrict__ Wvt,
                      bf16* __restrict__ Qb,   // [H*B][S][D]
                      bf16* __restrict__ Kb,   // [H*B][S][D]
                      bf16* __restrict__ Vt) { // [H*B][D][S]
  __shared__ bf16 btile[2][64 * KS_];          // 2 x 16 KB
  const int tidx  = threadIdx.x;
  const int lane  = tidx & 31;
  const int wave  = tidx >> 5;
  const int stile = blockIdx.x;
  const int hb    = blockIdx.y;
  const int h     = hb / B_;
  const int b     = hb % B_;
  const int which = blockIdx.z;
  const bf16* Wt  = (which == 0) ? Wqt : (which == 1) ? Wkt : Wvt;

  const int row = stile * 64 + wave * 16 + (lane & 15);
  const bf16* Abase = xb + ((size_t)b * S_ + row) * E_;
  const bf16* Bsrc  = Wt + (size_t)h * D_ * E_;   // 64 rows (d), stride E_

  stage_B(&btile[0][0], Bsrc, E_, tidx, wave);

  v8f c[4] = {};
  const int NST = E_ / KS_;
  for (int s = 0; s < NST; ++s) {
    const int ks = s * KS_;
    const int buf = s & 1;
    if (s + 1 < NST)
      stage_B(&btile[buf ^ 1][0], Bsrc + ks + KS_, E_, tidx, wave);

    v16bf av[4];
#pragma unroll
    for (int kk = 0; kk < 4; ++kk) av[kk] = loadA(Abase + ks + kk * 32, lane);

    stage_wait(wave, s + 1 < NST);
    __syncthreads();

#pragma unroll
    for (int kk = 0; kk < 4; ++kk) {
#pragma unroll
      for (int j = 0; j < 4; ++j) {
        const bf16* p = &btile[buf][(j * 16 + (lane & 15)) * KS_ + kk * 32];
        c[j] = wmma_bf16(av[kk], loadB(p, lane), c[j]);
      }
    }
    __syncthreads();
  }

  const int mbase = stile * 64 + wave * 16 + 8 * (lane >> 4);
  if (which == 2) {
    bf16* o = Vt + (size_t)hb * D_ * S_;
#pragma unroll
    for (int j = 0; j < 4; ++j) {
      int d = j * 16 + (lane & 15);
#pragma unroll
      for (int r = 0; r < 8; ++r)
        o[(size_t)d * S_ + (mbase + r)] = (bf16)c[j][r];
    }
  } else {
    bf16* o = ((which == 0) ? Qb : Kb) + (size_t)hb * S_ * D_;
#pragma unroll
    for (int j = 0; j < 4; ++j) {
      int d = j * 16 + (lane & 15);
#pragma unroll
      for (int r = 0; r < 8; ++r)
        o[(size_t)(mbase + r) * D_ + d] = (bf16)c[j][r];
    }
  }
}

// ---- scores = Q K^T / sqrt(D): per (h,b), GEMM M=S, N=S, K=64 --------------
// grid (S/64, S/64, H*B), block 128. Store-bound (1 GB f32 out); direct loads.

__global__ void k_scores(const bf16* __restrict__ Qb, const bf16* __restrict__ Kb,
                         float* __restrict__ wei) {   // [H*B][S][S]
  const int lane = threadIdx.x & 31;
  const int wave = threadIdx.x >> 5;
  const int mt = blockIdx.x, nt = blockIdx.y, hb = blockIdx.z;

  const int row = mt * 64 + wave * 16 + (lane & 15);
  const bf16* Abase = Qb + ((size_t)hb * S_ + row) * D_;
  const bf16* Bl    = Kb + ((size_t)hb * S_ + nt * 64 + (lane & 15)) * D_;

  v8f c[4] = {};
#pragma unroll
  for (int kc = 0; kc < D_; kc += 32) {
    v16bf a  = loadA(Abase + kc, lane);
    v16bf b0 = loadB(Bl + kc, lane);
    v16bf b1 = loadB(Bl + kc + (size_t)16 * D_, lane);
    v16bf b2 = loadB(Bl + kc + (size_t)32 * D_, lane);
    v16bf b3 = loadB(Bl + kc + (size_t)48 * D_, lane);
    c[0] = wmma_bf16(a, b0, c[0]);
    c[1] = wmma_bf16(a, b1, c[1]);
    c[2] = wmma_bf16(a, b2, c[2]);
    c[3] = wmma_bf16(a, b3, c[3]);
  }

  float* o = wei + (size_t)hb * S_ * S_;
  const int mbase = mt * 64 + wave * 16 + 8 * (lane >> 4);
#pragma unroll
  for (int j = 0; j < 4; ++j) {
    int t = nt * 64 + j * 16 + (lane & 15);
#pragma unroll
    for (int r = 0; r < 8; ++r)
      o[(size_t)(mbase + r) * S_ + t] = c[j][r] * 0.125f;  // 1/sqrt(64)
  }
}

// ---- softmax over rows of wei (in place, f32) ------------------------------

__global__ void k_softmax(float* __restrict__ wei) {
  __shared__ float red[256];
  const int t = threadIdx.x;
  float* row = wei + (size_t)blockIdx.x * S_;

  float v[8];
  float mx = -3.0e38f;
#pragma unroll
  for (int i = 0; i < 8; ++i) {
    v[i] = row[t + i * 256];
    mx = fmaxf(mx, v[i]);
  }
  red[t] = mx;
  __syncthreads();
  for (int s = 128; s > 0; s >>= 1) {
    if (t < s) red[t] = fmaxf(red[t], red[t + s]);
    __syncthreads();
  }
  mx = red[0];
  __syncthreads();

  float sum = 0.0f;
#pragma unroll
  for (int i = 0; i < 8; ++i) {
    v[i] = __expf(v[i] - mx);
    sum += v[i];
  }
  red[t] = sum;
  __syncthreads();
  for (int s = 128; s > 0; s >>= 1) {
    if (t < s) red[t] += red[t + s];
    __syncthreads();
  }
  const float inv = 1.0f / red[0];
#pragma unroll
  for (int i = 0; i < 8; ++i) row[t + i * 256] = v[i] * inv;
}

// ---- head_out = wei @ V: per (h,b), GEMM M=S, N=64, K=S --------------------
// grid (S/64, H*B), block 128. HBM-streaming A (wei, prefetched); B (Vt)
// staged in LDS via TDM.

__global__ void k_av(const float* __restrict__ wei, const bf16* __restrict__ Vt,
                     bf16* __restrict__ catb) {
  __shared__ bf16 btile[2][64 * KS_];
  const int tidx = threadIdx.x;
  const int lane = tidx & 31;
  const int wave = tidx >> 5;
  const int mt = blockIdx.x, hb = blockIdx.y;
  const int h = hb / B_;
  const int b = hb % B_;

  const int row = mt * 64 + wave * 16 + (lane & 15);
  const float* Abase = wei + ((size_t)hb * S_ + row) * S_;
  const bf16* Bsrc   = Vt + (size_t)hb * D_ * S_;   // 64 rows (d), stride S_

  stage_B(&btile[0][0], Bsrc, S_, tidx, wave);

  v8f c[4] = {};
  const int NST = S_ / KS_;
  for (int s = 0; s < NST; ++s) {
    const int ks = s * KS_;
    const int buf = s & 1;
    if (s + 1 < NST)
      stage_B(&btile[buf ^ 1][0], Bsrc + ks + KS_, S_, tidx, wave);

    __builtin_prefetch(Abase + ks + 2 * KS_, 0, 1);   // wei stream (HBM)
    v16bf av[4];
#pragma unroll
    for (int kk = 0; kk < 4; ++kk) av[kk] = loadA_f32(Abase + ks + kk * 32, lane);

    stage_wait(wave, s + 1 < NST);
    __syncthreads();

#pragma unroll
    for (int kk = 0; kk < 4; ++kk) {
#pragma unroll
      for (int j = 0; j < 4; ++j) {
        const bf16* p = &btile[buf][(j * 16 + (lane & 15)) * KS_ + kk * 32];
        c[j] = wmma_bf16(av[kk], loadB(p, lane), c[j]);
      }
    }
    __syncthreads();
  }

  const int mbase = mt * 64 + wave * 16 + 8 * (lane >> 4);
#pragma unroll
  for (int j = 0; j < 4; ++j) {
    int d = j * 16 + (lane & 15);
#pragma unroll
    for (int r = 0; r < 8; ++r)
      catb[((size_t)b * S_ + mbase + r) * HD_ + h * 64 + d] = (bf16)c[j][r];
  }
}

// ---- out = cat @ Wo + bo: GEMM M=B*S, N=E, K=HD ----------------------------
// grid (B*S/64, E/64), block 128. Wo tiles staged in LDS via TDM.

__global__ void k_proj(const bf16* __restrict__ catb, const bf16* __restrict__ Wot,
                       const float* __restrict__ bo, float* __restrict__ out) {
  __shared__ bf16 btile[2][64 * KS_];
  const int tidx = threadIdx.x;
  const int lane = tidx & 31;
  const int wave = tidx >> 5;
  const int mt = blockIdx.x, nt = blockIdx.y;

  const int row = mt * 64 + wave * 16 + (lane & 15);
  const bf16* Abase = catb + (size_t)row * HD_;
  const bf16* Bsrc  = Wot + (size_t)(nt * 64) * HD_;  // 64 rows (e), stride HD_

  stage_B(&btile[0][0], Bsrc, HD_, tidx, wave);

  v8f c[4] = {};
  const int NST = HD_ / KS_;
  for (int s = 0; s < NST; ++s) {
    const int ks = s * KS_;
    const int buf = s & 1;
    if (s + 1 < NST)
      stage_B(&btile[buf ^ 1][0], Bsrc + ks + KS_, HD_, tidx, wave);

    v16bf av[4];
#pragma unroll
    for (int kk = 0; kk < 4; ++kk) av[kk] = loadA(Abase + ks + kk * 32, lane);

    stage_wait(wave, s + 1 < NST);
    __syncthreads();

#pragma unroll
    for (int kk = 0; kk < 4; ++kk) {
#pragma unroll
      for (int j = 0; j < 4; ++j) {
        const bf16* p = &btile[buf][(j * 16 + (lane & 15)) * KS_ + kk * 32];
        c[j] = wmma_bf16(av[kk], loadB(p, lane), c[j]);
      }
    }
    __syncthreads();
  }

  const int mbase = mt * 64 + wave * 16 + 8 * (lane >> 4);
#pragma unroll
  for (int j = 0; j < 4; ++j) {
    int e = nt * 64 + j * 16 + (lane & 15);
#pragma unroll
    for (int r = 0; r < 8; ++r)
      out[(size_t)(mbase + r) * E_ + e] = c[j][r] + bo[e];
  }
}

// ---------------------------------------------------------------------------

extern "C" void kernel_launch(void* const* d_in, const int* in_sizes, int n_in,
                              void* d_out, int out_size, void* d_ws, size_t ws_size,
                              hipStream_t stream) {
  (void)in_sizes; (void)n_in; (void)out_size; (void)ws_size;

  const float* x  = (const float*)d_in[0];
  const float* Wq = (const float*)d_in[1];
  const float* Wk = (const float*)d_in[2];
  const float* Wv = (const float*)d_in[3];
  const float* Wo = (const float*)d_in[4];
  const float* bo = (const float*)d_in[5];

  float* out = (float*)d_out;                    // [B,S,E]
  float* wei = out + (size_t)B_ * S_ * E_;       // [H,B,S,S]

  char* ws = (char*)d_ws;                        // 88 MB total
  bf16* xb   = (bf16*)(ws);                      // [B,S,E]      16 MB
  bf16* Wqt  = (bf16*)(ws + (16u << 20));        // [H,D,E]       2 MB
  bf16* Wkt  = (bf16*)(ws + (18u << 20));        //               2 MB
  bf16* Wvt  = (bf16*)(ws + (20u << 20));        //               2 MB
  bf16* Wot  = (bf16*)(ws + (22u << 20));        // [E,HD]        2 MB
  bf16* Qb   = (bf16*)(ws + (24u << 20));        // [H*B,S,D]    16 MB
  bf16* Kb   = (bf16*)(ws + (40u << 20));        // [H*B,S,D]    16 MB
  bf16* Vt   = (bf16*)(ws + (56u << 20));        // [H*B,D,S]    16 MB
  bf16* catb = (bf16*)(ws + (72u << 20));        // [B,S,HD]     16 MB

  k_cvt_x<<<(B_ * S_ * E_) / 256, 256, 0, stream>>>(x, xb, B_ * S_ * E_);
  k_cvt_w_qkv<<<(H_ * E_ * D_) / 256, 256, 0, stream>>>(Wq, Wqt);
  k_cvt_w_qkv<<<(H_ * E_ * D_) / 256, 256, 0, stream>>>(Wk, Wkt);
  k_cvt_w_qkv<<<(H_ * E_ * D_) / 256, 256, 0, stream>>>(Wv, Wvt);
  k_cvt_wo<<<(HD_ * E_) / 256, 256, 0, stream>>>(Wo, Wot);

  k_qkv<<<dim3(S_ / 64, H_ * B_, 3), 128, 0, stream>>>(xb, Wqt, Wkt, Wvt, Qb, Kb, Vt);
  k_scores<<<dim3(S_ / 64, S_ / 64, H_ * B_), 128, 0, stream>>>(Qb, Kb, wei);
  k_softmax<<<H_ * B_ * S_, 256, 0, stream>>>(wei);
  k_av<<<dim3(S_ / 64, H_ * B_), 128, 0, stream>>>(wei, Vt, catb);
  k_proj<<<dim3(B_ * S_ / 64, E_ / 64), 128, 0, stream>>>(catb, Wot, bo, out);
}
